// CaptioningRNN_24867860644471
// MI455X (gfx1250) — matrix operand order repaired and checked by reading, not compiled, not run on hardware
//
#include <hip/hip_runtime.h>
#include <hip/hip_bf16.h>
#include <math.h>

// ---------------- Types for WMMA fragments (wave32, gfx1250) ----------------
typedef __attribute__((ext_vector_type(16))) _Float16 v16h;
typedef __attribute__((ext_vector_type(8)))  _Float16 v8h;
typedef __attribute__((ext_vector_type(8)))  float    v8f;
typedef __attribute__((ext_vector_type(2)))  float    v2f;

// Dimensions (fixed by the reference)
#define NN      128
#define T_STEPS 30
#define C_IN    1280
#define W_DIM   512
#define H_DIM   1024
#define G_DIM   4096      // 4*H
#define K_CAT   2560      // W_DIM + H_DIM + H_DIM
#define VOCAB   10000
#define T_CAP   31

// ---------------------------------------------------------------------------
// Fragment loaders for V_WMMA_F32_16X16X32_F16 (A: MxK row-major f16,
// B supplied pre-transposed as Bt[N][K] row-major f16).
// A layout (ISA 7.12.2, 16-bit A 16x32): lane L<16 -> M=L, halves v[0..7]=K0..7,
// v[8..15]=K16..23 ; lane L>=16 -> v[0..7]=K8..15, v[8..15]=K24..31.
// B layout (32x16): same K order as A: lanes 0-15 v[0..7]=K0..7, v[8..15]=K16..23; lanes 16-31 v[0..7]=K8..15, v[8..15]=K24..31.
// Works for both global and LDS-resident A (addrspace inferred -> ds_load_b128).
// ---------------------------------------------------------------------------
__device__ __forceinline__ v16h load_afrag_f16(const _Float16* A, int lda,
                                               int m0, int k0, int lane) {
  const _Float16* p = A + (size_t)(m0 + (lane & 15)) * lda + k0 + ((lane >> 4) << 3);
  v8h lo = *(const v8h*)(p);
  v8h hi = *(const v8h*)(p + 16);
  v16h a;
#pragma unroll
  for (int i = 0; i < 8; ++i) { a[i] = lo[i]; a[i + 8] = hi[i]; }
  return a;
}

__device__ __forceinline__ v16h load_bfrag_f16(const _Float16* __restrict__ Bt, int ldb,
                                               int n0, int k0, int lane) {
  const _Float16* p = Bt + (size_t)(n0 + (lane & 15)) * ldb + k0 + ((lane >> 4) << 3);
  v8h lo = *(const v8h*)(p);
  v8h hi = *(const v8h*)(p + 16);
  v16h b;
#pragma unroll
  for (int i = 0; i < 8; ++i) { b[i] = lo[i]; b[i + 8] = hi[i]; }
  return b;
}

// ---------------------------------------------------------------------------
// Async global->LDS tile staging (CDNA5 ASYNCcnt path; 16 B per lane issue).
// `halves` must be divisible by 8*nthreads so every lane runs the same trip
// count (EXEC all-ones). Caller must __syncthreads() afterwards.
// ---------------------------------------------------------------------------
__device__ __forceinline__ void async_stage_tile(_Float16* dstLds,
                                                 const _Float16* srcGlobal,
                                                 int halves, int tid, int nthreads) {
  for (int i = tid; i < halves / 8; i += nthreads) {
    unsigned lds_addr = (unsigned)(size_t)(dstLds + (size_t)i * 8);
    const void* gaddr = (const void*)(srcGlobal + (size_t)i * 8);
    asm volatile("global_load_async_to_lds_b128 %0, %1, off"
                 :: "v"(lds_addr), "v"(gaddr) : "memory");
  }
  asm volatile("s_wait_asynccnt 0x0" ::: "memory");
}

// ---------------------------------------------------------------------------
// Weight prep kernels (fp32 -> f16, transposed for contiguous B-fragments)
// ---------------------------------------------------------------------------
__global__ void build_wcat(const float* __restrict__ Wx, const float* __restrict__ Wh,
                           const float* __restrict__ Wattn, _Float16* __restrict__ Wcat) {
  int idx = blockIdx.x * blockDim.x + threadIdx.x;           // j*K_CAT + k
  if (idx >= G_DIM * K_CAT) return;
  int j = idx / K_CAT, k = idx - j * K_CAT;
  float v;
  if (k < W_DIM)              v = Wx[(size_t)k * G_DIM + j];
  else if (k < W_DIM + H_DIM) v = Wh[(size_t)(k - W_DIM) * G_DIM + j];
  else                        v = Wattn[(size_t)(k - W_DIM - H_DIM) * G_DIM + j];
  Wcat[idx] = (_Float16)v;
}

__global__ void build_wvocabT(const float* __restrict__ Wv, _Float16* __restrict__ WvT) {
  int idx = blockIdx.x * blockDim.x + threadIdx.x;           // v*H_DIM + h
  if (idx >= VOCAB * H_DIM) return;
  int v = idx >> 10, h = idx & (H_DIM - 1);
  WvT[idx] = (_Float16)Wv[(size_t)h * VOCAB + v];
}

__global__ void embed_gather(const int* __restrict__ captions, const float* __restrict__ Wemb,
                             _Float16* __restrict__ xs) {
  int idx = blockIdx.x * blockDim.x + threadIdx.x;           // ((t*NN+n)*W_DIM + w)
  if (idx >= T_STEPS * NN * W_DIM) return;
  int w = idx & (W_DIM - 1);
  int r = idx >> 9;
  int n = r & (NN - 1);
  int t = r >> 7;
  int tok = captions[n * T_CAP + t];                          // cap_in = captions[:, :-1]
  xs[idx] = (_Float16)Wemb[(size_t)tok * W_DIM + w];
}

// ---------------------------------------------------------------------------
// Projection GEMM in fp32 WMMA (16x16x4):  A[m=(n,p)][c]=images[n][c][p],
// B=W_proj[c][h], C=Apf[n][p][h] (+b_proj). M=2048, K=1280, N=1024.
// 1x4 register blocking: each wave computes a 16x64 strip (A-frag reused 4x).
// ---------------------------------------------------------------------------
__global__ __launch_bounds__(256) void proj_gemm_f32(const float* __restrict__ images,
                                                     const float* __restrict__ Wproj,
                                                     const float* __restrict__ bproj,
                                                     float* __restrict__ Apf) {
  int wave = blockIdx.x * (blockDim.x >> 5) + (threadIdx.x >> 5);
  int lane = threadIdx.x & 31;
  int tm = wave >> 4, q = wave & 15;                          // 128 row tiles x 16 quads
  int m0 = tm << 4, n0 = q << 6;
  int mrow = m0 + (lane & 15);
  int nimg = mrow >> 4, p = mrow & 15;
  const float* imgbase = images + (size_t)nimg * C_IN * 16 + p;
  int klane = (lane >> 4) << 1;
  int coll = lane & 15;
  v8f acc[4] = {v8f{}, v8f{}, v8f{}, v8f{}};
  for (int k0 = 0; k0 < C_IN; k0 += 4) {
    int k = k0 + klane;
    v2f a;
    a.x = imgbase[(size_t)k * 16];
    a.y = imgbase[(size_t)(k + 1) * 16];
    const float* w0 = Wproj + (size_t)k * H_DIM + n0 + coll;
    const float* w1 = Wproj + (size_t)(k + 1) * H_DIM + n0 + coll;
#pragma unroll
    for (int j = 0; j < 4; ++j) {
      v2f b;
      b.x = w0[j << 4];
      b.y = w1[j << 4];
      acc[j] = __builtin_amdgcn_wmma_f32_16x16x4_f32(false, a, false, b, (short)0, acc[j],
                                                     false, false);
    }
  }
  int rbase = m0 + ((lane >> 4) << 3);
#pragma unroll
  for (int j = 0; j < 4; ++j) {
    int col = n0 + (j << 4) + coll;
    float bv = bproj[col];
#pragma unroll
    for (int r = 0; r < 8; ++r)
      Apf[(size_t)(rbase + r) * H_DIM + col] = acc[j][r] + bv;
  }
}

__global__ void h0_kernel(const float* __restrict__ Apf, float* __restrict__ hbuf,
                          float* __restrict__ cbuf) {
  int idx = blockIdx.x * blockDim.x + threadIdx.x;            // n*H + h
  if (idx >= NN * H_DIM) return;
  int n = idx >> 10;
  int hh = idx & (H_DIM - 1);
  const float* base = Apf + (size_t)n * 16 * H_DIM + hh;
  float s = 0.f;
#pragma unroll
  for (int p = 0; p < 16; ++p) s += base[p * H_DIM];
  s *= (1.0f / 16.0f);
  hbuf[idx] = s;
  cbuf[idx] = s;
}

// ---------------------------------------------------------------------------
// Per-step attention + build concatenated f16 activation row [x_t | h | attn]
// ---------------------------------------------------------------------------
__global__ __launch_bounds__(256) void attn_cat(const float* __restrict__ Apf,
                                                const float* __restrict__ hbuf,
                                                const _Float16* __restrict__ xs_t,
                                                _Float16* __restrict__ cat) {
  __shared__ float sh_h[H_DIM];
  __shared__ float sc[16];
  __shared__ float se[16];
  int n = blockIdx.x, tid = threadIdx.x;                      // 256 threads
  const float* hrow = hbuf + (size_t)n * H_DIM;
  const float* Arow = Apf + (size_t)n * 16 * H_DIM;
  for (int i = tid; i < H_DIM; i += 256) sh_h[i] = hrow[i];
  __syncthreads();
  // scores[p] = (h . A[:,p]) / 32 ; 16 groups of 16 lanes
  int p = tid >> 4, l = tid & 15;
  const float* Ap = Arow + p * H_DIM;
  float partial = 0.f;
  for (int i = l; i < H_DIM; i += 16) partial += sh_h[i] * Ap[i];
#pragma unroll
  for (int m = 8; m; m >>= 1) partial += __shfl_xor(partial, m, 16);
  if (l == 0) sc[p] = partial * (1.0f / 32.0f);
  __syncthreads();
  if (tid < 16) {
    float mx = sc[0];
#pragma unroll
    for (int i = 1; i < 16; ++i) mx = fmaxf(mx, sc[i]);
    se[tid] = expf(sc[tid] - mx);
  }
  __syncthreads();
  float denom = 0.f;
#pragma unroll
  for (int i = 0; i < 16; ++i) denom += se[i];
  float inv = 1.0f / denom;

  _Float16* crow = cat + (size_t)n * K_CAT;
  const _Float16* xrow = xs_t + (size_t)n * W_DIM;
  for (int i = tid; i < W_DIM; i += 256) crow[i] = xrow[i];
  for (int i = tid; i < H_DIM; i += 256) {
    crow[W_DIM + i] = (_Float16)sh_h[i];
    float a = 0.f;
#pragma unroll
    for (int pp = 0; pp < 16; ++pp) a += se[pp] * Arow[pp * H_DIM + i];
    crow[W_DIM + H_DIM + i] = (_Float16)(a * inv);
  }
}

// ---------------------------------------------------------------------------
// Gates GEMM (f16 WMMA): C[128][4096] = cat[128][2560] * Wcat[4096][2560]^T + b
// 64 blocks x 8 waves; all waves of a block share one 16-row A tile, staged in
// LDS (80 KB) via async global->LDS; each wave computes a 16x64 strip.
// ---------------------------------------------------------------------------
__global__ __launch_bounds__(256) void gates_gemm(const _Float16* __restrict__ A,
                                                  const _Float16* __restrict__ Bt,
                                                  const float* __restrict__ bias,
                                                  float* __restrict__ C) {
  __shared__ _Float16 sA[16 * K_CAT];                         // 80 KB of the WGP's 320 KB
  int tid = threadIdx.x;
  int lane = tid & 31, wv = tid >> 5;
  int tm = blockIdx.x >> 3;                                   // 8 blocks per row tile
  int m0 = tm << 4;
  async_stage_tile(sA, A + (size_t)m0 * K_CAT, 16 * K_CAT, tid, 256);  // 20 issues/lane
  __syncthreads();
  int q = ((blockIdx.x & 7) << 3) + wv;                       // 0..63
  int n0 = q << 6;                                            // 16x64 strip
  v8f acc[4] = {v8f{}, v8f{}, v8f{}, v8f{}};
  for (int k0 = 0; k0 < K_CAT; k0 += 32) {
    if (k0 + 32 < K_CAT)
      __builtin_prefetch(Bt + (size_t)(n0 + (lane & 15)) * K_CAT + k0 + 32, 0, 1);
    v16h a = load_afrag_f16(sA, K_CAT, 0, k0, lane);
#pragma unroll
    for (int j = 0; j < 4; ++j) {
      v16h b = load_bfrag_f16(Bt, K_CAT, n0 + (j << 4), k0, lane);
      acc[j] = __builtin_amdgcn_wmma_f32_16x16x32_f16(false, a, false, b, (short)0, acc[j],
                                                      false, false);
    }
  }
  int coll = lane & 15;
  int rbase = m0 + ((lane >> 4) << 3);
#pragma unroll
  for (int j = 0; j < 4; ++j) {
    int col = n0 + (j << 4) + coll;
    float bv = bias[col];
#pragma unroll
    for (int r = 0; r < 8; ++r)
      C[(size_t)(rbase + r) * G_DIM + col] = acc[j][r] + bv;
  }
}

// ---------------------------------------------------------------------------
// LSTM gate nonlinearity + state update; also emits f16 h for vocab GEMM
// ---------------------------------------------------------------------------
__global__ void lstm_update(const float* __restrict__ gates, float* __restrict__ hbuf,
                            float* __restrict__ cbuf, _Float16* __restrict__ hs_t) {
  int idx = blockIdx.x * blockDim.x + threadIdx.x;            // n*H + h
  if (idx >= NN * H_DIM) return;
  int n = idx >> 10;
  int hh = idx & (H_DIM - 1);
  const float* g = gates + (size_t)n * G_DIM;
  float gi = g[hh], gf = g[H_DIM + hh], go = g[2 * H_DIM + hh], gg = g[3 * H_DIM + hh];
  float si = 1.0f / (1.0f + expf(-gi));
  float sf = 1.0f / (1.0f + expf(-gf));
  float so = 1.0f / (1.0f + expf(-go));
  float tg = tanhf(gg);
  float cn = sf * cbuf[idx] + si * tg;
  float hn = so * tanhf(cn);
  cbuf[idx] = cn;
  hbuf[idx] = hn;
  hs_t[idx] = (_Float16)hn;
}

// ---------------------------------------------------------------------------
// Vocab GEMM fused with online log-softmax + masked NLL.
// Block = 16 rows (m = t*128+n) with the hs tile staged in LDS (32 KB, async);
// 8 waves stream the 625 column tiles as 156 quads + 1 tail tile.
// ---------------------------------------------------------------------------
__device__ __forceinline__ void tile_stats(const v8f& acc, int n0, int lane,
                                           const float* __restrict__ bvocab,
                                           const int* tgtcol, float* rmax, float* rsum,
                                           float* tval) {
  int col = n0 + (lane & 15);
  float bv = bvocab[col];
#pragma unroll
  for (int r = 0; r < 8; ++r) {
    float v = acc[r] + bv;
    if (col == tgtcol[r]) tval[r] += v;
    float tm = v;
#pragma unroll
    for (int m = 8; m; m >>= 1) tm = fmaxf(tm, __shfl_xor(tm, m, 16));
    float ts = expf(v - tm);
#pragma unroll
    for (int m = 8; m; m >>= 1) ts += __shfl_xor(ts, m, 16);
    float nm = fmaxf(rmax[r], tm);
    rsum[r] = rsum[r] * expf(rmax[r] - nm) + ts * expf(tm - nm);
    rmax[r] = nm;
  }
}

__global__ __launch_bounds__(256) void vocab_nll(const _Float16* __restrict__ hs,
                                                 const _Float16* __restrict__ WvT,
                                                 const float* __restrict__ bvocab,
                                                 const int* __restrict__ captions,
                                                 float* __restrict__ partials) {
  __shared__ _Float16 sA[16 * H_DIM];                         // 32 KB
  __shared__ float smax[8][16];
  __shared__ float ssum[8][16];
  __shared__ float stgt[8][16];
  int tid = threadIdx.x;
  int lane = tid & 31, wv = tid >> 5;
  int half = lane >> 4;
  int m0 = blockIdx.x << 4;
  async_stage_tile(sA, hs + (size_t)m0 * H_DIM, 16 * H_DIM, tid, 256);  // 8 issues/lane
  __syncthreads();

  int tgtcol[8];
#pragma unroll
  for (int r = 0; r < 8; ++r) {
    int row = m0 + r + (half << 3);
    int t = row >> 7, n = row & (NN - 1);
    tgtcol[r] = captions[n * T_CAP + t + 1];                  // cap_out
  }
  float rmax[8], rsum[8], tval[8];
#pragma unroll
  for (int r = 0; r < 8; ++r) { rmax[r] = -1e30f; rsum[r] = 0.f; tval[r] = 0.f; }

  // 156 full quads (tiles 0..623)
  for (int q = wv; q < 156; q += 8) {
    int n0 = q << 6;
    v8f acc[4] = {v8f{}, v8f{}, v8f{}, v8f{}};
    for (int k0 = 0; k0 < H_DIM; k0 += 32) {
      v16h a = load_afrag_f16(sA, H_DIM, 0, k0, lane);
#pragma unroll
      for (int j = 0; j < 4; ++j) {
        v16h b = load_bfrag_f16(WvT, H_DIM, n0 + (j << 4), k0, lane);
        acc[j] = __builtin_amdgcn_wmma_f32_16x16x32_f16(false, a, false, b, (short)0, acc[j],
                                                        false, false);
      }
    }
#pragma unroll
    for (int j = 0; j < 4; ++j)
      tile_stats(acc[j], n0 + (j << 4), lane, bvocab, tgtcol, rmax, rsum, tval);
  }
  // tail tile 624 (cols 9984..9999), handled by wave 4 (= 156 % 8)
  if (wv == 4) {
    int n0 = 624 << 4;
    v8f acc = {};
    for (int k0 = 0; k0 < H_DIM; k0 += 32) {
      v16h a = load_afrag_f16(sA, H_DIM, 0, k0, lane);
      v16h b = load_bfrag_f16(WvT, H_DIM, n0, k0, lane);
      acc = __builtin_amdgcn_wmma_f32_16x16x32_f16(false, a, false, b, (short)0, acc,
                                                   false, false);
    }
    tile_stats(acc, n0, lane, bvocab, tgtcol, rmax, rsum, tval);
  }

#pragma unroll
  for (int r = 0; r < 8; ++r) {
#pragma unroll
    for (int m = 8; m; m >>= 1) tval[r] += __shfl_xor(tval[r], m, 16);
  }
  if ((lane & 15) == 0) {
#pragma unroll
    for (int r = 0; r < 8; ++r) {
      int lr = r + (half << 3);
      smax[wv][lr] = rmax[r];
      ssum[wv][lr] = rsum[r];
      stgt[wv][lr] = tval[r];
    }
  }
  __syncthreads();
  if (tid < 16) {
    int lr = tid;
    float mx = smax[0][lr];
#pragma unroll
    for (int w2 = 1; w2 < 8; ++w2) mx = fmaxf(mx, smax[w2][lr]);
    float s = 0.f, tv = 0.f;
#pragma unroll
    for (int w2 = 0; w2 < 8; ++w2) {
      s += ssum[w2][lr] * expf(smax[w2][lr] - mx);
      tv += stgt[w2][lr];
    }
    int row = m0 + lr;
    int t = row >> 7, n = row & (NN - 1);
    int tgt = captions[n * T_CAP + t + 1];
    smax[0][lr] = (tgt != 0) ? (mx + logf(s) - tv) : 0.0f;    // masked nll
  }
  __syncthreads();
  if (tid == 0) {
    float s = 0.f;
#pragma unroll
    for (int i = 0; i < 16; ++i) s += smax[0][i];
    partials[blockIdx.x] = s;
  }
}

__global__ void finalize_loss(const float* __restrict__ partials, float* __restrict__ out) {
  if (threadIdx.x == 0 && blockIdx.x == 0) {
    float s = 0.f;
    for (int i = 0; i < (T_STEPS * NN) / 16; ++i) s += partials[i];
    out[0] = s * (1.0f / (float)NN);
  }
}

// ---------------------------------------------------------------------------
extern "C" void kernel_launch(void* const* d_in, const int* in_sizes, int n_in,
                              void* d_out, int out_size, void* d_ws, size_t ws_size,
                              hipStream_t stream) {
  (void)in_sizes; (void)n_in; (void)out_size; (void)ws_size;
  const float* images   = (const float*)d_in[0];
  const int*   captions = (const int*)d_in[1];
  const float* W_embed  = (const float*)d_in[2];
  const float* W_proj   = (const float*)d_in[3];
  const float* b_proj   = (const float*)d_in[4];
  const float* Wx       = (const float*)d_in[5];
  const float* Wh       = (const float*)d_in[6];
  const float* Wattn    = (const float*)d_in[7];
  const float* bgate    = (const float*)d_in[8];
  const float* W_vocab  = (const float*)d_in[9];
  const float* b_vocab  = (const float*)d_in[10];
  float* out = (float*)d_out;

  char* ws = (char*)d_ws;
  size_t off = 0;
  auto alloc = [&](size_t bytes) -> void* {
    void* p = (void*)(ws + off);
    off = (off + bytes + 255) & ~(size_t)255;
    return p;
  };
  _Float16* Wcat  = (_Float16*)alloc((size_t)G_DIM * K_CAT * 2);        // 21.0 MB
  _Float16* WvT   = (_Float16*)alloc((size_t)VOCAB * H_DIM * 2);        // 20.5 MB
  _Float16* xs    = (_Float16*)alloc((size_t)T_STEPS * NN * W_DIM * 2); // 3.9 MB
  float*    Apf   = (float*)alloc((size_t)NN * 16 * H_DIM * 4);         // 8.4 MB
  float*    hbuf  = (float*)alloc((size_t)NN * H_DIM * 4);
  float*    cbuf  = (float*)alloc((size_t)NN * H_DIM * 4);
  _Float16* cat   = (_Float16*)alloc((size_t)NN * K_CAT * 2);
  float*    gates = (float*)alloc((size_t)NN * G_DIM * 4);
  _Float16* hs    = (_Float16*)alloc((size_t)T_STEPS * NN * H_DIM * 2); // 7.9 MB
  float*    parts = (float*)alloc(1024);

  // --- weight / input prep ---
  build_wcat<<<(G_DIM * K_CAT) / 256, 256, 0, stream>>>(Wx, Wh, Wattn, Wcat);
  build_wvocabT<<<(VOCAB * H_DIM) / 256, 256, 0, stream>>>(W_vocab, WvT);
  embed_gather<<<(T_STEPS * NN * W_DIM) / 256, 256, 0, stream>>>(captions, W_embed, xs);

  // --- image projection (fp32 WMMA, 16x64 per wave) + h0 ---
  {
    int waves = (NN * 16 / 16) * (H_DIM / 64);   // 2048 waves
    proj_gemm_f32<<<waves / 8, 256, 0, stream>>>(images, W_proj, b_proj, Apf);
  }
  h0_kernel<<<(NN * H_DIM) / 256, 256, 0, stream>>>(Apf, hbuf, cbuf);

  // --- LSTM scan (sequential over 30 steps) ---
  for (int t = 0; t < T_STEPS; ++t) {
    attn_cat<<<NN, 256, 0, stream>>>(Apf, hbuf, xs + (size_t)t * NN * W_DIM, cat);
    gates_gemm<<<64, 256, 0, stream>>>(cat, Wcat, bgate, gates);
    lstm_update<<<(NN * H_DIM) / 256, 256, 0, stream>>>(gates, hbuf, cbuf,
                                                        hs + (size_t)t * NN * H_DIM);
  }

  // --- vocab GEMM fused with log-softmax NLL ---
  vocab_nll<<<(T_STEPS * NN) / 16, 256, 0, stream>>>(hs, WvT, b_vocab, captions, parts);
  finalize_loss<<<1, 64, 0, stream>>>(parts, out);
}
